// SelfAttention_64707977281815
// MI455X (gfx1250) — matrix operand order repaired and checked by reading, not compiled, not run on hardware
//
#include <hip/hip_runtime.h>

// ---------------------------------------------------------------------------
// Self-attention forward for MI455X (gfx1250), wave32 + WMMA bf16.
//   B=4, L=2048, C=1024, H=16, D=64.  SCALE = 1/sqrt(64)/4 = 0.03125.
// Pipeline: cvt(f32->bf16) -> QKV GEMM (fused split/scale/transpose-V)
//           -> flash attention (online softmax, bf16 WMMA, P via LDS)
//           -> out projection GEMM (+b_proj) -> f32.
// GEMMs use double-buffered LDS tiles staged with gfx1250 async-to-LDS
// (GLOBAL_LOAD_ASYNC_TO_LDS_B128 / ASYNCcnt) when the builtin is available.
// ---------------------------------------------------------------------------

#define L_    2048
#define C_    1024
#define H_    16
#define D_    64
#define NB_   4
#define ROWS_ (NB_ * L_)           // 8192
#define SCALE_ 0.03125f

typedef __attribute__((ext_vector_type(16))) __bf16 v16bf;
typedef __attribute__((ext_vector_type(8)))  float  v8f;
typedef __attribute__((ext_vector_type(4)))  int    v4i;

struct alignas(16) U4 { unsigned int x, y, z, w; };
struct U4x2 { U4 a, b; };

static __device__ __forceinline__ v16bf make_frag(U4 a, U4 b) {
    U4x2 u{a, b};
    return __builtin_bit_cast(v16bf, u);
}

// ----- async global->LDS staging (gfx1250), with portable fallback ---------
#if defined(__gfx1250__) && __has_builtin(__builtin_amdgcn_global_load_async_to_lds_b128)
#define HAVE_ASYNC 1
#else
#define HAVE_ASYNC 0
#endif

static __device__ __forceinline__ void cp16(const __bf16* g, __bf16* l) {
#if HAVE_ASYNC
    // Builtin wants (v4i AS1*, v4i AS3*, imm offset, imm cpol).
    // AS0->AS1 is bit-identical for global; AS0->AS3 is low-32 truncation.
    __builtin_amdgcn_global_load_async_to_lds_b128(
        (__attribute__((address_space(1))) v4i*)(unsigned long long)g,
        (__attribute__((address_space(3))) v4i*)(unsigned int)(unsigned long long)l,
        0, 0);
#else
    *(U4*)l = *(const U4*)g;
#endif
}

static __device__ __forceinline__ void wait_cp() {
#if HAVE_ASYNC
#if __has_builtin(__builtin_amdgcn_s_wait_asynccnt)
    __builtin_amdgcn_s_wait_asynccnt(0);
#else
    asm volatile("s_wait_asynccnt 0x0" ::: "memory");
#endif
#endif
}

static __device__ __forceinline__ float redmax16(float x) {
    x = fmaxf(x, __shfl_xor(x, 1));
    x = fmaxf(x, __shfl_xor(x, 2));
    x = fmaxf(x, __shfl_xor(x, 4));
    x = fmaxf(x, __shfl_xor(x, 8));
    return x;
}
static __device__ __forceinline__ float redsum16(float x) {
    x += __shfl_xor(x, 1);
    x += __shfl_xor(x, 2);
    x += __shfl_xor(x, 4);
    x += __shfl_xor(x, 8);
    return x;
}

// ---------------------------------------------------------------------------
// f32 -> bf16 elementwise convert
// ---------------------------------------------------------------------------
__global__ __launch_bounds__(256) void cvt_kernel(const float* __restrict__ in,
                                                  __bf16* __restrict__ out, int n) {
    int i = blockIdx.x * 256 + threadIdx.x;
    if (i < n) out[i] = (__bf16)in[i];
}

// ---------------------------------------------------------------------------
// 128x128 GEMM: C[row,col] = sum_k A[row,k]*Bm[col,k], K = 1024, both operands
// row-major with contraction contiguous (x @ W^T pattern).
// 8 waves: (wm 0..3) x (wn 0..1) -> 32x64 per wave = 2x4 16x16 f32 accums.
// Double-buffered 128x32 LDS tiles (rows padded to 40 bf16), async staged.
// ---------------------------------------------------------------------------
static __device__ __forceinline__ void stage_tiles(
    const __bf16* __restrict__ A, const __bf16* __restrict__ Bm,
    int row0, int col0, int k0, __bf16* sA, __bf16* sB, int t)
{
    // 512 16B-chunks per matrix (1024 per pair); thread t handles chunks t, t+256.
    const int r0 = t >> 2;
    const int ko = (t & 3) * 8;
    cp16(A  + (size_t)(row0 + r0)      * C_ + k0 + ko, sA + r0 * 40 + ko);
    cp16(Bm + (size_t)(col0 + r0)      * C_ + k0 + ko, sB + r0 * 40 + ko);
    cp16(A  + (size_t)(row0 + r0 + 64) * C_ + k0 + ko, sA + (r0 + 64) * 40 + ko);
    cp16(Bm + (size_t)(col0 + r0 + 64) * C_ + k0 + ko, sB + (r0 + 64) * 40 + ko);
}

static __device__ __forceinline__ void compute_step(
    const __bf16* sA, const __bf16* sB, v8f acc[2][4],
    int wm, int wn, int n16, int hi)
{
    v16bf af[2], bf2[4];
    #pragma unroll
    for (int mt = 0; mt < 2; ++mt) {
        // A-frag: lane m=n16 holds K = hi*8+j and 16+hi*8+j (j=0..7)
        const __bf16* p = sA + (wm * 32 + mt * 16 + n16) * 40;
        af[mt] = make_frag(*(const U4*)(p + hi * 8),
                           *(const U4*)(p + 16 + hi * 8));
    }
    #pragma unroll
    for (int nt = 0; nt < 4; ++nt) {
        // B-frag: lane n=n16 holds K = hi*8+j and 16+hi*8+j (j=0..7)
        const __bf16* p = sB + (wn * 64 + nt * 16 + n16) * 40;
        bf2[nt] = make_frag(*(const U4*)(p + hi * 8),
                            *(const U4*)(p + 16 + hi * 8));
    }
    #pragma unroll
    for (int mt = 0; mt < 2; ++mt)
        #pragma unroll
        for (int nt = 0; nt < 4; ++nt)
            acc[mt][nt] = __builtin_amdgcn_wmma_f32_16x16x32_bf16(
                false, af[mt], false, bf2[nt], (short)0, acc[mt][nt],
                false, false);
}

static __device__ __forceinline__ void gemm128_core(
    const __bf16* __restrict__ A, const __bf16* __restrict__ Bm,
    int row0, int col0,
    __bf16* sA0, __bf16* sB0, __bf16* sA1, __bf16* sB1, v8f acc[2][4])
{
    const int t    = threadIdx.x;
    const int lane = t & 31;
    const int wid  = t >> 5;
    const int wm   = wid & 3;
    const int wn   = wid >> 2;
    const int n16  = lane & 15;
    const int hi   = lane >> 4;

    stage_tiles(A, Bm, row0, col0, 0, sA0, sB0, t);
    wait_cp();
    __syncthreads();

    for (int k0 = 0; k0 < C_; k0 += 64) {
        // buffer 0 holds tile k0; prefetch k0+32 into buffer 1
        stage_tiles(A, Bm, row0, col0, k0 + 32, sA1, sB1, t);
        compute_step(sA0, sB0, acc, wm, wn, n16, hi);
        wait_cp();
        __syncthreads();

        if (k0 + 64 < C_)
            stage_tiles(A, Bm, row0, col0, k0 + 64, sA0, sB0, t);
        compute_step(sA1, sB1, acc, wm, wn, n16, hi);
        wait_cp();
        __syncthreads();
    }
}

// ---------------------------------------------------------------------------
// QKV projection: [8192,1024] x [3072,1024]^T -> split Q/K/V, fuse bias+scale.
// Q,K -> bf16 [B,H,L,D];  V -> bf16 TRANSPOSED [B,H,D,L] (for PV B-frags).
// grid (3072/128, 8192/128), block 256.
// ---------------------------------------------------------------------------
__global__ __launch_bounds__(256) void qkv_kernel(
    const __bf16* __restrict__ xb, const __bf16* __restrict__ wqkv,
    const float* __restrict__ q_bias, const float* __restrict__ v_bias,
    __bf16* __restrict__ Qb, __bf16* __restrict__ Kb, __bf16* __restrict__ Vt)
{
    __shared__ __bf16 sA0[128 * 40], sB0[128 * 40];
    __shared__ __bf16 sA1[128 * 40], sB1[128 * 40];
    const int row0 = blockIdx.y * 128;
    const int col0 = blockIdx.x * 128;
    const int lane = threadIdx.x & 31;
    const int wid  = threadIdx.x >> 5;
    const int wm = wid & 3, wn = wid >> 2;
    const int n16 = lane & 15, hi = lane >> 4;

    v8f acc[2][4];
    const v8f vzero = {0.f, 0.f, 0.f, 0.f, 0.f, 0.f, 0.f, 0.f};
    #pragma unroll
    for (int mt = 0; mt < 2; ++mt)
        #pragma unroll
        for (int nt = 0; nt < 4; ++nt) acc[mt][nt] = vzero;

    gemm128_core(xb, wqkv, row0, col0, sA0, sB0, sA1, sB1, acc);

    #pragma unroll
    for (int mt = 0; mt < 2; ++mt) {
        #pragma unroll
        for (int nt = 0; nt < 4; ++nt) {
            const int col = col0 + wn * 64 + nt * 16 + n16;
            const int seg = col >> 10;            // 0=q 1=k 2=v
            const int cc  = col & 1023;
            const int h   = cc >> 6;
            const int d   = cc & 63;
            #pragma unroll
            for (int r = 0; r < 8; ++r) {
                const int row = row0 + wm * 32 + mt * 16 + hi * 8 + r;
                const int b = row >> 11, l = row & 2047;
                float v = acc[mt][nt][r];
                if (seg == 0) {
                    Qb[((size_t)(b * H_ + h) * L_ + l) * D_ + d] =
                        (__bf16)((v + q_bias[cc]) * SCALE_);
                } else if (seg == 1) {
                    Kb[((size_t)(b * H_ + h) * L_ + l) * D_ + d] = (__bf16)v;
                } else {
                    Vt[((size_t)(b * H_ + h) * D_ + d) * L_ + l] =
                        (__bf16)(v + v_bias[cc]);
                }
            }
        }
    }
}

// ---------------------------------------------------------------------------
// Flash attention. grid (L/128, H, B), block 256 (8 waves).
// Each wave owns 16 q-rows, streams all 2048 k-rows in chunks of 32.
// S-tiles: 2x (16x16) via 2 WMMAs each over D=64. Online softmax in f32.
// P (C-layout) bounced through 1KB/wave LDS to become an A-frag (16x32 bf16),
// then O += P @ V via 4 WMMAs against transposed V.
// ---------------------------------------------------------------------------
__global__ __launch_bounds__(256) void attn_kernel(
    const __bf16* __restrict__ Qb, const __bf16* __restrict__ Kb,
    const __bf16* __restrict__ Vt, const float* __restrict__ bias,
    __bf16* __restrict__ Ob)
{
    __shared__ __bf16 sP[8][512];                 // 16x32 bf16 per wave
    const int lane = threadIdx.x & 31;
    const int w    = threadIdx.x >> 5;
    const int n16  = lane & 15, hi = lane >> 4;
    const int h = blockIdx.y, b = blockIdx.z;
    const int bh = b * H_ + h;
    const int qbase = blockIdx.x * 128 + w * 16;

    const __bf16* Qp = Qb + ((size_t)bh * L_ + qbase) * D_;
    const __bf16* Kp = Kb + (size_t)bh * L_ * D_;
    const __bf16* Vp = Vt + (size_t)bh * D_ * L_;

    // Preload Q fragments for this wave's 16 rows (2 chunks over D=64).
    v16bf qf[2];
    #pragma unroll
    for (int kk = 0; kk < 2; ++kk) {
        const __bf16* p = Qp + (size_t)n16 * D_ + kk * 32;
        qf[kk] = make_frag(*(const U4*)(p + hi * 8),
                           *(const U4*)(p + 16 + hi * 8));
    }

    const v8f vzero = {0.f, 0.f, 0.f, 0.f, 0.f, 0.f, 0.f, 0.f};
    v8f o[4] = {vzero, vzero, vzero, vzero};      // 16q x 64d output accum
    float rmax[8], rsum[8];
    #pragma unroll
    for (int r = 0; r < 8; ++r) { rmax[r] = -1e30f; rsum[r] = 0.f; }

    for (int kb = 0; kb < L_; kb += 32) {
        v8f s0 = vzero, s1 = vzero;               // cols kb..+15 / kb+16..+31
        #pragma unroll
        for (int kk = 0; kk < 2; ++kk) {
            const __bf16* k0p = Kp + (size_t)(kb + n16) * D_ + kk * 32;
            const __bf16* k1p = Kp + (size_t)(kb + 16 + n16) * D_ + kk * 32;
            v16bf b0 = make_frag(*(const U4*)(k0p + hi * 8),
                                 *(const U4*)(k0p + 16 + hi * 8));
            v16bf b1 = make_frag(*(const U4*)(k1p + hi * 8),
                                 *(const U4*)(k1p + 16 + hi * 8));
            s0 = __builtin_amdgcn_wmma_f32_16x16x32_bf16(false, qf[kk], false,
                     b0, (short)0, s0, false, false);
            s1 = __builtin_amdgcn_wmma_f32_16x16x32_bf16(false, qf[kk], false,
                     b1, (short)0, s1, false, false);
        }

        // Add attention bias; online softmax update.
        const float* bp = bias + (size_t)qbase * L_ + kb;
        float alpha[8];
        #pragma unroll
        for (int r = 0; r < 8; ++r) {
            const int m = hi * 8 + r;
            s0[r] += bp[(size_t)m * L_ + n16];
            s1[r] += bp[(size_t)m * L_ + 16 + n16];
            float tm = redmax16(fmaxf(s0[r], s1[r]));
            float mn = fmaxf(rmax[r], tm);
            alpha[r] = __expf(rmax[r] - mn);
            rmax[r]  = mn;
            float p0 = __expf(s0[r] - mn);
            float p1 = __expf(s1[r] - mn);
            s0[r] = p0; s1[r] = p1;
            rsum[r] = rsum[r] * alpha[r] + redsum16(p0 + p1);
            #pragma unroll
            for (int tt = 0; tt < 4; ++tt) o[tt][r] *= alpha[r];
        }

        // C-layout P -> LDS -> A-frag layout (16x32 bf16).
        __bf16* pw = &sP[w][0];
        #pragma unroll
        for (int r = 0; r < 8; ++r) {
            const int m = hi * 8 + r;
            pw[m * 32 + n16]      = (__bf16)s0[r];
            pw[m * 32 + 16 + n16] = (__bf16)s1[r];
        }
        const __bf16* pr = &sP[w][n16 * 32];
        v16bf ap = make_frag(*(const U4*)(pr + hi * 8),
                             *(const U4*)(pr + 16 + hi * 8));

        // O += P @ V : B-frags from transposed V are two 16B runs per lane.
        #pragma unroll
        for (int tt = 0; tt < 4; ++tt) {
            const __bf16* vp = Vp + (size_t)(tt * 16 + n16) * L_ + kb;
            v16bf bv = make_frag(*(const U4*)(vp + hi * 8),
                                 *(const U4*)(vp + 16 + hi * 8));
            o[tt] = __builtin_amdgcn_wmma_f32_16x16x32_bf16(false, ap, false,
                        bv, (short)0, o[tt], false, false);
        }
    }

    // Normalize and store to [B, L, C] bf16 buffer.
    #pragma unroll
    for (int r = 0; r < 8; ++r) {
        const float inv = 1.0f / rsum[r];
        const int qrow = qbase + hi * 8 + r;
        #pragma unroll
        for (int tt = 0; tt < 4; ++tt) {
            Ob[(size_t)(b * L_ + qrow) * C_ + h * 64 + tt * 16 + n16] =
                (__bf16)(o[tt][r] * inv);
        }
    }
}

// ---------------------------------------------------------------------------
// Output projection: [8192,1024] x [1024,1024]^T + b_proj -> f32 d_out.
// grid (1024/128, 8192/128), block 256.
// ---------------------------------------------------------------------------
__global__ __launch_bounds__(256) void proj_kernel(
    const __bf16* __restrict__ Ob, const __bf16* __restrict__ wproj,
    const float* __restrict__ b_proj, float* __restrict__ out)
{
    __shared__ __bf16 sA0[128 * 40], sB0[128 * 40];
    __shared__ __bf16 sA1[128 * 40], sB1[128 * 40];
    const int row0 = blockIdx.y * 128;
    const int col0 = blockIdx.x * 128;
    const int lane = threadIdx.x & 31;
    const int wid  = threadIdx.x >> 5;
    const int wm = wid & 3, wn = wid >> 2;
    const int n16 = lane & 15, hi = lane >> 4;

    v8f acc[2][4];
    const v8f vzero = {0.f, 0.f, 0.f, 0.f, 0.f, 0.f, 0.f, 0.f};
    #pragma unroll
    for (int mt = 0; mt < 2; ++mt)
        #pragma unroll
        for (int nt = 0; nt < 4; ++nt) acc[mt][nt] = vzero;

    gemm128_core(Ob, wproj, row0, col0, sA0, sB0, sA1, sB1, acc);

    #pragma unroll
    for (int mt = 0; mt < 2; ++mt) {
        #pragma unroll
        for (int nt = 0; nt < 4; ++nt) {
            const int col = col0 + wn * 64 + nt * 16 + n16;
            const float bb = b_proj[col];
            #pragma unroll
            for (int r = 0; r < 8; ++r) {
                const int row = row0 + wm * 32 + mt * 16 + hi * 8 + r;
                out[(size_t)row * C_ + col] = acc[mt][nt][r] + bb;
            }
        }
    }
}

// ---------------------------------------------------------------------------
// Host-side launch. Inputs: x, attn_bias, W_qkv, q_bias, v_bias, W_proj, b_proj.
// Workspace layout (bf16, ~88MB total):
//   xb[8192*1024] | wqkv[3072*1024] | wproj[1024*1024]
//   | Q[B,H,L,D] | K[B,H,L,D] | Vt[B,H,D,L] | O[B*L, C]
// ---------------------------------------------------------------------------
extern "C" void kernel_launch(void* const* d_in, const int* in_sizes, int n_in,
                              void* d_out, int out_size, void* d_ws, size_t ws_size,
                              hipStream_t stream) {
    (void)in_sizes; (void)n_in; (void)out_size; (void)ws_size;
    const float* x         = (const float*)d_in[0];
    const float* attn_bias = (const float*)d_in[1];
    const float* W_qkv     = (const float*)d_in[2];
    const float* q_bias    = (const float*)d_in[3];
    const float* v_bias    = (const float*)d_in[4];
    const float* W_proj    = (const float*)d_in[5];
    const float* b_proj    = (const float*)d_in[6];
    float* out = (float*)d_out;

    __bf16* xb    = (__bf16*)d_ws;
    __bf16* wqkvb = xb + (size_t)ROWS_ * C_;
    __bf16* wprjb = wqkvb + (size_t)3 * C_ * C_;
    __bf16* Qb    = wprjb + (size_t)C_ * C_;
    __bf16* Kb    = Qb + (size_t)ROWS_ * C_;
    __bf16* Vt    = Kb + (size_t)ROWS_ * C_;
    __bf16* Ob    = Vt + (size_t)ROWS_ * C_;

    cvt_kernel<<<(ROWS_ * C_) / 256, 256, 0, stream>>>(x, xb, ROWS_ * C_);
    cvt_kernel<<<(3 * C_ * C_) / 256, 256, 0, stream>>>(W_qkv, wqkvb, 3 * C_ * C_);
    cvt_kernel<<<(C_ * C_) / 256, 256, 0, stream>>>(W_proj, wprjb, C_ * C_);

    qkv_kernel<<<dim3(3 * C_ / 128, ROWS_ / 128), 256, 0, stream>>>(
        xb, wqkvb, q_bias, v_bias, Qb, Kb, Vt);

    attn_kernel<<<dim3(L_ / 128, H_, NB_), 256, 0, stream>>>(
        Qb, Kb, Vt, attn_bias, Ob);

    proj_kernel<<<dim3(C_ / 128, ROWS_ / 128), 256, 0, stream>>>(
        Ob, wprjb, b_proj, out);
}
